// PixelDifferenceConvolution_76390288327657
// MI455X (gfx1250) — matrix operand order repaired and checked by reading, not compiled, not run on hardware
//
#include <hip/hip_runtime.h>
#include <hip/hip_bf16.h>

// ---------------------------------------------------------------------------
// Pixel-difference conv as implicit GEMM with bf16 WMMA (gfx1250 / CDNA5).
//   M = C_OUT = 128, K = C_IN*9 = 576 (tap-major: K = tap*64 + ci), N = pixels.
//   D = A(bf16 weights, center tap folded) x B(bf16 im2col staged in LDS) + C
//   via V_WMMA_F32_16X16X32_BF16 with f32 accumulation.
//   sched_group_barrier enforces a depth-2 pipeline: [DS8(s+1)][WMMA4(s)].
// ---------------------------------------------------------------------------

#define C_IN   64
#define C_OUT  128
#define HW     224
#define NTILE  64          // pixels (w) per block
#define KSTEPS 18          // 576 / 32
#define NB     16          // batch
#define LDS_CI 72          // padded ci row (bf16): stride 36 dwords -> conflict-free b128

#if defined(__has_builtin)
#if __has_builtin(__builtin_amdgcn_sched_group_barrier)
#define SCHED_GROUP(mask, size, id) __builtin_amdgcn_sched_group_barrier((mask), (size), (id))
#endif
#endif
#ifndef SCHED_GROUP
#define SCHED_GROUP(mask, size, id)
#endif

#define SG_DS_READ 0x100
#define SG_WMMA    0x008

typedef __attribute__((ext_vector_type(16))) __bf16 v16bf;
typedef __attribute__((ext_vector_type(8)))  float  v8f;

union Frag {
    uint4 q[2];
    v16bf v;
};

__device__ __forceinline__ unsigned short f32_to_bf16_rne(float f) {
    unsigned u = __float_as_uint(f);
    unsigned r = (u + 0x7FFFu + ((u >> 16) & 1u)) >> 16;
    return (unsigned short)r;
}

__device__ __forceinline__ int refl224(int i) {
    if (i < 0)    i = -i;
    if (i >= HW)  i = 2 * HW - 2 - i;
    return i;
}

// ---------------------------------------------------------------------------
// Kernel 1: W_full in bf16, pre-swizzled into 16-bit A fragments with contiguous k
// (k = s*32 + 16*(lane/16) + 2v + j).  Storage (dwords): addr = ((mtile*18 + s)*32 + lane)*8 + v
// ---------------------------------------------------------------------------
__global__ void pdc_prep_weights(const float* __restrict__ weight,
                                 unsigned* __restrict__ wsA) {
    int t = blockIdx.x * blockDim.x + threadIdx.x;
    if (t >= 8 * KSTEPS * 32 * 8) return;

    int v    = t & 7;
    int lane = (t >> 3) & 31;
    int rest = t >> 8;            // mtile*18 + s
    int s     = rest % KSTEPS;
    int mtile = rest / KSTEPS;

    int m  = lane & 15;
    int co = mtile * 16 + m;

    int khalf = (v >= 4) ? 16 : 0;
    int koff  = (lane >= 16) ? 8 : 0;
    int k0    = s * 32 + 2 * koff + 2 * v;

    float vals[2];
#pragma unroll
    for (int j = 0; j < 2; ++j) {
        int K   = k0 + j;
        int tap = K >> 6;         // 0..8 over the 3x3 window
        int ci  = K & 63;
        float w;
        if (tap == 4) {           // derived center tap = -sum(neighbors)
            float sum = 0.f;
#pragma unroll
            for (int q = 0; q < 8; ++q)
                sum += weight[(size_t)co * 512 + ci * 8 + q];
            w = -sum;
        } else {
            int ti = (tap > 4) ? (tap - 1) : tap;
            w = weight[(size_t)co * 512 + ci * 8 + ti];
        }
        vals[j] = w;
    }
    unsigned lo = f32_to_bf16_rne(vals[0]);
    unsigned hi = f32_to_bf16_rne(vals[1]);
    wsA[t] = lo | (hi << 16);
}

// ---------------------------------------------------------------------------
// Fragment load helpers
// ---------------------------------------------------------------------------
__device__ __forceinline__ void loadA(const unsigned* __restrict__ aBase,
                                      int s, Frag& a) {
    const uint4* ap = (const uint4*)(aBase + (size_t)s * 256);
    a.q[0] = ap[0];
    a.q[1] = ap[1];
}

__device__ __forceinline__ void loadB(const unsigned short xs[3][NTILE + 2][LDS_CI],
                                      int s, int klo, int laneM, Frag* b) {
    int tap = s >> 1;                       // 2 K-steps per tap
    int cb  = ((s & 1) << 5) + klo;         // ci base for this lane
    int dyi = (tap >= 6) ? 2 : ((tap >= 3) ? 1 : 0);
    int dxi = tap - dyi * 3;                // == dx + 1
    const unsigned short* bRow = &xs[dyi][0][cb];
#pragma unroll
    for (int t = 0; t < 4; ++t) {
        int col = t * 16 + laneM + dxi;     // (pixel + dx + 1)
        const uint4* bp = (const uint4*)(bRow + (size_t)col * LDS_CI);
        b[t].q[0] = bp[0];
        b[t].q[1] = bp[1];
    }
}

// ---------------------------------------------------------------------------
// Kernel 2: conv.  Block = (n, h, 64-pixel w tile), 256 threads = 8 waves.
// LDS x-halo: xs[3 rows][66 w][72 ci pad] bf16.  Wave w owns channels
// [16w,16w+16) x 64 pixels: 4 accumulators, 18 K-steps, 72 WMMAs.
// ---------------------------------------------------------------------------
__global__ void __launch_bounds__(256, 2)
pdc_conv(const float* __restrict__ x,
         const unsigned* __restrict__ wsA,
         const float* __restrict__ bias,
         float* __restrict__ out) {
    __shared__ __align__(16) unsigned short xs[3][NTILE + 2][LDS_CI];

    int bid  = blockIdx.x;
    int wblk = bid & 3;
    int q    = bid >> 2;
    int h    = q % HW;
    int n    = q / HW;
    int w0   = wblk * NTILE;

    int tid = threadIdx.x;

    // ---- stage x halo (3 rows x 66 cols x 64 ci) into LDS as bf16 ----
    if (tid < 3 * C_IN) {
        int dy = tid >> 6;          // 0..2
        int ci = tid & 63;
        int row = refl224(h + dy - 1);
        const float* src = x + (((size_t)n * C_IN + ci) * HW + row) * HW;

        // reflect edges (w0-1 and w0+64)
        xs[dy][0][ci]         = f32_to_bf16_rne(src[refl224(w0 - 1)]);
        xs[dy][NTILE + 1][ci] = f32_to_bf16_rne(src[refl224(w0 + NTILE)]);

        if (w0 + NTILE <= HW) {
            // full tile: 16 aligned float4 loads
            const float4* s4 = (const float4*)(src + w0);
#pragma unroll
            for (int c = 0; c < NTILE / 4; ++c) {
                float4 f = s4[c];
                int jb = 1 + c * 4;
                xs[dy][jb + 0][ci] = f32_to_bf16_rne(f.x);
                xs[dy][jb + 1][ci] = f32_to_bf16_rne(f.y);
                xs[dy][jb + 2][ci] = f32_to_bf16_rne(f.z);
                xs[dy][jb + 3][ci] = f32_to_bf16_rne(f.w);
            }
        } else {
            // partial last tile: per-element reflect
            for (int j = 1; j <= NTILE; ++j)
                xs[dy][j][ci] = f32_to_bf16_rne(src[refl224(w0 - 1 + j)]);
        }
    }
    __syncthreads();

    int wave  = tid >> 5;
    int lane  = tid & 31;
    int laneM = lane & 15;
    int hi    = (lane >> 4) & 1;    // lane half selects K-half / M-half
    int klo   = hi << 4;            // B fragment: ci offset 0 or 16

    v8f acc[4];
#pragma unroll
    for (int t = 0; t < 4; ++t) acc[t] = (v8f)(0.0f);

    // A fragments at ((wave*18 + s)*32 + lane)*8 dwords
    const unsigned* aBase = wsA + (((size_t)wave * KSTEPS) * 32 + lane) * 8;

    // ---- software-pipelined K loop, schedule enforced by group barriers:
    //      DS8(0) | DS8(1) WMMA4(0) | DS8(2) WMMA4(1) | ... | WMMA4(17)
    Frag a0;
    Frag b0[4];
    loadA(aBase, 0, a0);
    loadB(xs, 0, klo, laneM, b0);
    SCHED_GROUP(SG_DS_READ, 8, 0);      // group: prologue B loads (s=0)

#pragma unroll
    for (int s = 0; s < KSTEPS; ++s) {
        Frag a1;
        Frag b1[4];
        if (s + 1 < KSTEPS) {
            loadA(aBase, s + 1, a1);
            loadB(xs, s + 1, klo, laneM, b1);
        }

#pragma unroll
        for (int t = 0; t < 4; ++t) {
            acc[t] = __builtin_amdgcn_wmma_f32_16x16x32_bf16(
                false, a0.v, false, b0[t].v, (short)0, acc[t], false, false);
        }

        if (s + 1 < KSTEPS) {
            SCHED_GROUP(SG_DS_READ, 8, 0);   // group: B loads for s+1
        }
        SCHED_GROUP(SG_WMMA, 4, 0);          // group: WMMAs for s

        if (s + 1 < KSTEPS) {
            a0 = a1;
#pragma unroll
            for (int t = 0; t < 4; ++t) b0[t] = b1[t];
        }
    }

    // ---- epilogue: C/D layout -> VGPR r is M=r (lanes 0-15) / M=8+r ----
    int coB = wave * 16 + hi * 8;
    float bv[8];
#pragma unroll
    for (int r = 0; r < 8; ++r) bv[r] = bias[coB + r];

#pragma unroll
    for (int t = 0; t < 4; ++t) {
        int w = w0 + t * 16 + laneM;
        if (w < HW) {
#pragma unroll
            for (int r = 0; r < 8; ++r) {
                size_t o = (((size_t)n * C_OUT + coB + r) * HW + h) * HW + w;
                out[o] = acc[t][r] + bv[r];
            }
        }
    }
}

// ---------------------------------------------------------------------------
extern "C" void kernel_launch(void* const* d_in, const int* in_sizes, int n_in,
                              void* d_out, int out_size, void* d_ws, size_t ws_size,
                              hipStream_t stream) {
    const float* x      = (const float*)d_in[0];
    const float* weight = (const float*)d_in[1];
    const float* bias   = (const float*)d_in[2];
    float*       out    = (float*)d_out;
    unsigned*    wsA    = (unsigned*)d_ws;   // needs 8*18*32*8*4 = 147456 B

    int prepThreads = 8 * KSTEPS * 32 * 8;   // 36864
    pdc_prep_weights<<<(prepThreads + 255) / 256, 256, 0, stream>>>(weight, wsA);

    int blocks = NB * HW * 4;                // 14336
    pdc_conv<<<blocks, 256, 0, stream>>>(x, wsA, bias, out);
}
